// SimpleClawMatrix_52424370815699
// MI455X (gfx1250) — matrix operand order repaired and checked by reading, not compiled, not run on hardware
//
#include <hip/hip_runtime.h>

#define DEV __device__ __forceinline__

typedef __bf16 bf16_t;
typedef __attribute__((ext_vector_type(16))) __bf16 v16bf;
typedef __attribute__((ext_vector_type(8)))  __bf16 v8bf;
typedef __attribute__((ext_vector_type(8)))  float  v8f;
typedef unsigned int u32x4 __attribute__((ext_vector_type(4)));
typedef int i32x4 __attribute__((ext_vector_type(4)));
typedef int i32x8 __attribute__((ext_vector_type(8)));

constexpr int kB = 16;
constexpr int kS = 2048;
constexpr int kD = 768;
constexpr float kInvTemp = 1.0f / 0.07f;
constexpr float kLnEps = 1e-5f;
constexpr int kLdsLd = 68;   // 64-elem rows + 4-elem pad (TDM pad: 2 dwords / 32 dwords)

#if __has_builtin(__builtin_amdgcn_tensor_load_to_lds) && \
    __has_builtin(__builtin_amdgcn_s_wait_tensorcnt)
#define USE_TDM 1
#else
#define USE_TDM 0
#endif

// ---- helpers -------------------------------------------------------------

DEV bf16_t f2bf(float f) {
  unsigned u = __builtin_bit_cast(unsigned, f);
  u += 0x7fffu + ((u >> 16) & 1u);               // round-to-nearest-even
  unsigned short h = (unsigned short)(u >> 16);
  return __builtin_bit_cast(bf16_t, h);
}

DEV v8f wmma_bf16(v16bf a, v16bf b, v8f c) {
  return __builtin_amdgcn_wmma_f32_16x16x32_bf16(false, a, false, b,
                                                 (short)0, c, false, false);
}

// A-fragment 16x32 (MxK) bf16 from row-major source, stride lda (elements).
DEV v16bf load_a_bf(const bf16_t* a, int lda) {
  const int lane = (int)(threadIdx.x & 31);
  const int m = lane & 15, sel = lane >> 4;
  const bf16_t* p = a + (size_t)m * lda + sel * 16;
  v8bf lo = *(const v8bf*)p;
  v8bf hi = *(const v8bf*)(p + 8);
  v16bf r;
#pragma unroll
  for (int i = 0; i < 8; ++i) { r[i] = lo[i]; r[i + 8] = hi[i]; }
  return r;
}

// B-fragment 32x16 (KxN) from row-major B^T [N x K], stride ldb.
DEV v16bf load_b_bf(const bf16_t* bt, int ldb) {
  const int lane = (int)(threadIdx.x & 31);
  const int n = lane & 15, sel = lane >> 4;
  return *(const v16bf*)(bt + (size_t)n * ldb + sel * 16);
}

// B-fragment 32x16 gathered (transposed) from a row-major [32 x 64] LDS tile.
DEV v16bf gather_b_tr(const bf16_t* t, int n0) {
  const int lane = (int)(threadIdx.x & 31);
  const int n = n0 + (lane & 15);
  const int ks = (lane >> 4) * 16;
  v16bf r;
#pragma unroll
  for (int j = 0; j < 16; ++j) r[j] = t[(ks + j) * kLdsLd + n];
  return r;
}

// A-fragment 16x32 gathered (transposed) from a row-major [32 x 64] LDS tile.
DEV v16bf gather_a_tr(const bf16_t* t, int m0) {
  const int lane = (int)(threadIdx.x & 31);
  const int m = m0 + (lane & 15);
  const int sel = lane >> 4;
  v16bf r;
#pragma unroll
  for (int j = 0; j < 8; ++j) {
    r[j]     = t[(sel * 16 + j) * kLdsLd + m];
    r[j + 8] = t[(sel * 16 + 8 + j) * kLdsLd + m];
  }
  return r;
}

// Fallback staging: copy a row-major 32x64 bf16 tile global->LDS (128 thr).
DEV void stage_rowmajor(bf16_t* dst, const bf16_t* src, int srcLd) {
#pragma unroll
  for (int j = 0; j < 2; ++j) {
    const int idx = ((int)threadIdx.x * 2 + j) * 8;  // 0..2040
    const int kk = idx >> 6, dd = idx & 63;
    v8bf v = *(const v8bf*)(src + (size_t)kk * srcLd + dd);
#pragma unroll
    for (int i = 0; i < 8; ++i) dst[kk * kLdsLd + dd + i] = v[i];
  }
}

#if USE_TDM
// TDM: async DMA of a 2D bf16 tile (64 x 32 rows) global -> LDS, with
// LDS padding of 2 dwords after every 32 dwords (row stride 68 elems).
DEV void tdm_load_2d(const bf16_t* gsrc, unsigned lds_off,
                     long long rowStrideElems) {
  const unsigned long long ga = (unsigned long long)(size_t)gsrc;
  u32x4 g0;
  g0[0] = 1u;                                          // count=1
  g0[1] = lds_off;                                     // lds_addr
  g0[2] = (unsigned)(ga & 0xffffffffu);                // global_addr lo
  g0[3] = (unsigned)((ga >> 32) & 0x01ffffffu) | (2u << 30);  // addr hi | type=2
  const unsigned long long s0 = (unsigned long long)rowStrideElems;
  i32x8 g1;
  g1[0] = (int)0x03110000u;        // data_size=2B, pad_en, interval=32dw, amt=2dw
  g1[1] = (int)(64u << 16);        // tensor_dim0 = 64
  g1[2] = (int)(32u << 16);        // tensor_dim1 = 32
  g1[3] = (int)(64u << 16);        // tile_dim0 = 64
  g1[4] = (int)32u;                // tile_dim1 = 32, tile_dim2 = 0
  g1[5] = (int)(unsigned)(s0 & 0xffffffffu);           // dim0 stride lo
  g1[6] = (int)(unsigned)((s0 >> 32) & 0xffffu);       // dim0 stride hi
  g1[7] = 0;
  i32x4 gz = {0, 0, 0, 0};
#if __clang_major__ >= 23
  i32x8 gz8 = {0, 0, 0, 0, 0, 0, 0, 0};
  __builtin_amdgcn_tensor_load_to_lds(g0, g1, gz, gz, gz8, 0);
#else
  __builtin_amdgcn_tensor_load_to_lds(g0, g1, gz, gz, 0);
#endif
}
#endif

// ---- kernel 1: f32 -> bf16 conversion ------------------------------------

__global__ void k_f32_to_bf16(const float* __restrict__ src,
                              bf16_t* __restrict__ dst, int n) {
  int i = blockIdx.x * blockDim.x + threadIdx.x;
  if (i < n) dst[i] = f2bf(src[i]);
}

// ---- kernel 2: projections  P = X @ W^T + b  (bf16 in/out) ---------------
// grid(384, B, 2), block 128 (4 waves). Each wave: 16x64 tile, K=768.
// Software-pipelined: next k-step's fragments load while current WMMAs run.

__global__ void k_proj(const bf16_t* __restrict__ Xv, const bf16_t* __restrict__ Xl,
                       const bf16_t* __restrict__ Wv, const bf16_t* __restrict__ Wl,
                       const float* __restrict__ bv, const float* __restrict__ bl,
                       bf16_t* __restrict__ Pv, bf16_t* __restrict__ Pl) {
  const int wave = (int)(threadIdx.x >> 5);
  const int lane = (int)(threadIdx.x & 31);
  const int tile = blockIdx.x * 4 + wave;
  const int mt = tile & 127;
  const int nt = tile >> 7;
  const int b = blockIdx.y;
  const bf16_t* X; const bf16_t* W; const float* bias; bf16_t* P;
  if (blockIdx.z == 0) { X = Xv; W = Wv; bias = bv; P = Pv; }
  else                 { X = Xl; W = Wl; bias = bl; P = Pl; }
  X += (size_t)b * kS * kD;
  P += (size_t)b * kS * kD;
  const int m0 = mt * 16, n0 = nt * 64;
  const bf16_t* Xr = X + (size_t)m0 * kD;

  v8f acc[4] = {};
  v16bf a_cur = load_a_bf(Xr, kD);
  v16bf b_cur[4];
#pragma unroll
  for (int t = 0; t < 4; ++t)
    b_cur[t] = load_b_bf(W + (size_t)(n0 + t * 16) * kD, kD);

  for (int k = 0; k < kD; k += 32) {
    const bool more = (k + 32) < kD;
    v16bf a_nxt = a_cur;
    v16bf b_nxt[4];
    if (more) {
      a_nxt = load_a_bf(Xr + k + 32, kD);
#pragma unroll
      for (int t = 0; t < 4; ++t)
        b_nxt[t] = load_b_bf(W + (size_t)(n0 + t * 16) * kD + k + 32, kD);
    }
#pragma unroll
    for (int t = 0; t < 4; ++t) acc[t] = wmma_bf16(a_cur, b_cur[t], acc[t]);
    a_cur = a_nxt;
    if (more) {
#pragma unroll
      for (int t = 0; t < 4; ++t) b_cur[t] = b_nxt[t];
    }
  }
  const int nl = lane & 15, rbase = (lane >> 4) * 8;
#pragma unroll
  for (int t = 0; t < 4; ++t) {
    const int n = n0 + t * 16 + nl;
    const float bsv = bias[n];
#pragma unroll
    for (int r = 0; r < 8; ++r)
      P[(size_t)(m0 + rbase + r) * kD + n] = f2bf(acc[t][r] + bsv);
  }
}

// ---- kernel 3: sim = vp @ lp^T / T, softmax, attn bf16 out ---------------
// Group-pipelined B-fragment loads (4 tiles ahead) against WMMA bursts.

__global__ void k_sim_softmax(const bf16_t* __restrict__ vp,
                              const bf16_t* __restrict__ lp,
                              bf16_t* __restrict__ attn) {
  const int b = blockIdx.y;
  const int m0 = blockIdx.x * 16;
  const int wave = (int)(threadIdx.x >> 5);
  const int lane = (int)(threadIdx.x & 31);
  const bf16_t* A  = vp + (size_t)b * kS * kD + (size_t)m0 * kD;
  const bf16_t* Bt = lp + (size_t)b * kS * kD + (size_t)(wave * 256) * kD;

  v8f acc[16] = {};
  for (int k = 0; k < kD; k += 32) {
    v16bf af = load_a_bf(A + k, kD);
    v16bf bcur[4];
#pragma unroll
    for (int u = 0; u < 4; ++u)
      bcur[u] = load_b_bf(Bt + (size_t)(u * 16) * kD + k, kD);
#pragma unroll
    for (int g = 0; g < 4; ++g) {
      v16bf bnxt[4];
      if (g < 3) {
#pragma unroll
        for (int u = 0; u < 4; ++u)
          bnxt[u] = load_b_bf(Bt + (size_t)(((g + 1) * 4 + u) * 16) * kD + k, kD);
      }
#pragma unroll
      for (int u = 0; u < 4; ++u)
        acc[g * 4 + u] = wmma_bf16(af, bcur[u], acc[g * 4 + u]);
      if (g < 3) {
#pragma unroll
        for (int u = 0; u < 4; ++u) bcur[u] = bnxt[u];
      }
    }
  }

  float mx[8];
#pragma unroll
  for (int r = 0; r < 8; ++r) mx[r] = -3.0e38f;
#pragma unroll
  for (int t = 0; t < 16; ++t)
#pragma unroll
    for (int r = 0; r < 8; ++r) {
      acc[t][r] *= kInvTemp;
      mx[r] = fmaxf(mx[r], acc[t][r]);
    }
#pragma unroll
  for (int r = 0; r < 8; ++r)
    for (int s = 1; s < 16; s <<= 1) mx[r] = fmaxf(mx[r], __shfl_xor(mx[r], s, 32));

  __shared__ float smax[8][16];
  __shared__ float ssum[8][16];
  if (lane == 0)  for (int r = 0; r < 8; ++r) smax[wave][r]     = mx[r];
  if (lane == 16) for (int r = 0; r < 8; ++r) smax[wave][8 + r] = mx[r];
  __syncthreads();

  const int rbase = (lane >> 4) * 8;
  float rowmax[8];
#pragma unroll
  for (int r = 0; r < 8; ++r) {
    float v = -3.0e38f;
    for (int w = 0; w < 8; ++w) v = fmaxf(v, smax[w][rbase + r]);
    rowmax[r] = v;
  }

  float sum[8] = {};
#pragma unroll
  for (int t = 0; t < 16; ++t)
#pragma unroll
    for (int r = 0; r < 8; ++r) {
      float e = __expf(acc[t][r] - rowmax[r]);
      acc[t][r] = e;
      sum[r] += e;
    }
#pragma unroll
  for (int r = 0; r < 8; ++r)
    for (int s = 1; s < 16; s <<= 1) sum[r] += __shfl_xor(sum[r], s, 32);
  if (lane == 0)  for (int r = 0; r < 8; ++r) ssum[wave][r]     = sum[r];
  if (lane == 16) for (int r = 0; r < 8; ++r) ssum[wave][8 + r] = sum[r];
  __syncthreads();

  float rs[8];
#pragma unroll
  for (int r = 0; r < 8; ++r) {
    float v = 0.f;
    for (int w = 0; w < 8; ++w) v += ssum[w][rbase + r];
    rs[r] = 1.0f / v;
  }

  bf16_t* O = attn + (size_t)b * kS * kS;
  const int nl = lane & 15;
#pragma unroll
  for (int t = 0; t < 16; ++t) {
    const int n = wave * 256 + t * 16 + nl;
#pragma unroll
    for (int r = 0; r < 8; ++r)
      O[(size_t)(m0 + rbase + r) * kS + n] = f2bf(acc[t][r] * rs[r]);
  }
}

// ---- kernel 4: aligned_vision = attn @ vp --------------------------------
// TDM (double-buffered) stages the vp [32k x 64d] tile; transpose on read.

__global__ void k_aligned_vision(const bf16_t* __restrict__ attn,
                                 const bf16_t* __restrict__ vp,
                                 bf16_t* __restrict__ av) {
  __shared__ bf16_t lds[2][32 * kLdsLd];
  const int b = blockIdx.y;
  const int qb = blockIdx.x & 31;
  const int db = blockIdx.x >> 5;
  const int wave = (int)(threadIdx.x >> 5);
  const int lane = (int)(threadIdx.x & 31);
  const int q0 = qb * 64, d0 = db * 64;
  const bf16_t* At = attn + (size_t)b * kS * kS + (size_t)(q0 + wave * 16) * kS;
  const bf16_t* Vb = vp + (size_t)b * kS * kD + d0;

#if USE_TDM
  if (wave == 0)
    tdm_load_2d(Vb, (unsigned)(size_t)&lds[0][0], kD);
#else
  stage_rowmajor(&lds[0][0], Vb, kD);
#endif

  v8f acc[4] = {};
  for (int k = 0; k < kS; k += 32) {
    const int cur = (k >> 5) & 1, nxt = cur ^ 1;
    const bool more = (k + 32) < kS;
    if (more) {
#if USE_TDM
      if (wave == 0)
        tdm_load_2d(Vb + (size_t)(k + 32) * kD, (unsigned)(size_t)&lds[nxt][0], kD);
#else
      stage_rowmajor(&lds[nxt][0], Vb + (size_t)(k + 32) * kD, kD);
#endif
    }
    __builtin_prefetch(At + k + 128, 0, 0);   // global_prefetch of A stream
#if USE_TDM
    if (wave == 0) {
      if (more) __builtin_amdgcn_s_wait_tensorcnt(1);
      else      __builtin_amdgcn_s_wait_tensorcnt(0);
    }
#endif
    __syncthreads();
    v16bf af = load_a_bf(At + k, kS);
    v16bf bt[4];
#pragma unroll
    for (int t = 0; t < 4; ++t) bt[t] = gather_b_tr(&lds[cur][0], t * 16);
#pragma unroll
    for (int t = 0; t < 4; ++t) acc[t] = wmma_bf16(af, bt[t], acc[t]);
    __syncthreads();
  }
  const int nl = lane & 15, rbase = (lane >> 4) * 8;
  bf16_t* ob = av + (size_t)b * kS * kD;
#pragma unroll
  for (int t = 0; t < 4; ++t) {
    const int n = d0 + t * 16 + nl;
#pragma unroll
    for (int r = 0; r < 8; ++r)
      ob[(size_t)(q0 + wave * 16 + rbase + r) * kD + n] = f2bf(acc[t][r]);
  }
}

// ---- kernel 5: aligned_language = attn^T @ lp ----------------------------
// Both operands column-strided: TDM-stage both tiles row-major, gather-T.

__global__ void k_aligned_language(const bf16_t* __restrict__ attn,
                                   const bf16_t* __restrict__ lp,
                                   bf16_t* __restrict__ al) {
  __shared__ bf16_t ldsA[2][32 * kLdsLd];
  __shared__ bf16_t ldsB[2][32 * kLdsLd];
  const int b = blockIdx.y;
  const int kb = blockIdx.x & 31;
  const int db = blockIdx.x >> 5;
  const int wave = (int)(threadIdx.x >> 5);
  const int lane = (int)(threadIdx.x & 31);
  const int kp0 = kb * 64, d0 = db * 64;
  const bf16_t* Ab = attn + (size_t)b * kS * kS + kp0;
  const bf16_t* Lb = lp + (size_t)b * kS * kD + d0;

#if USE_TDM
  if (wave == 0) {
    tdm_load_2d(Ab, (unsigned)(size_t)&ldsA[0][0], kS);
    tdm_load_2d(Lb, (unsigned)(size_t)&ldsB[0][0], kD);
  }
#else
  stage_rowmajor(&ldsA[0][0], Ab, kS);
  stage_rowmajor(&ldsB[0][0], Lb, kD);
#endif

  v8f acc[4] = {};
  for (int q = 0; q < kS; q += 32) {
    const int cur = (q >> 5) & 1, nxt = cur ^ 1;
    const bool more = (q + 32) < kS;
    if (more) {
#if USE_TDM
      if (wave == 0) {
        tdm_load_2d(Ab + (size_t)(q + 32) * kS, (unsigned)(size_t)&ldsA[nxt][0], kS);
        tdm_load_2d(Lb + (size_t)(q + 32) * kD, (unsigned)(size_t)&ldsB[nxt][0], kD);
      }
#else
      stage_rowmajor(&ldsA[nxt][0], Ab + (size_t)(q + 32) * kS, kS);
      stage_rowmajor(&ldsB[nxt][0], Lb + (size_t)(q + 32) * kD, kD);
#endif
    }
#if USE_TDM
    if (wave == 0) {
      if (more) __builtin_amdgcn_s_wait_tensorcnt(2);
      else      __builtin_amdgcn_s_wait_tensorcnt(0);
    }
#endif
    __syncthreads();
    v16bf af = gather_a_tr(&ldsA[cur][0], wave * 16);
    v16bf bt[4];
#pragma unroll
    for (int t = 0; t < 4; ++t) bt[t] = gather_b_tr(&ldsB[cur][0], t * 16);
#pragma unroll
    for (int t = 0; t < 4; ++t) acc[t] = wmma_bf16(af, bt[t], acc[t]);
    __syncthreads();
  }
  const int nl = lane & 15, rbase = (lane >> 4) * 8;
  bf16_t* ob = al + (size_t)b * kS * kD;
#pragma unroll
  for (int t = 0; t < 4; ++t) {
    const int n = d0 + t * 16 + nl;
#pragma unroll
    for (int r = 0; r < 8; ++r)
      ob[(size_t)(kp0 + wave * 16 + rbase + r) * kD + n] = f2bf(acc[t][r]);
  }
}

// ---- kernel 6: head GEMM + bias + LayerNorm + ReLU (f32 out) -------------
// Software-pipelined K loop like k_proj.

__global__ void k_head_ln_relu(const bf16_t* __restrict__ av,
                               const bf16_t* __restrict__ al,
                               const bf16_t* __restrict__ WoB,
                               const float* __restrict__ bo,
                               const float* __restrict__ gamma,
                               const float* __restrict__ beta,
                               float* __restrict__ out) {
  const int b = blockIdx.y;
  const int m0 = blockIdx.x * 16;
  const int wave = (int)(threadIdx.x >> 5);
  const int lane = (int)(threadIdx.x & 31);
  const bf16_t* avB = av + (size_t)b * kS * kD + (size_t)m0 * kD;
  const bf16_t* alB = al + (size_t)b * kS * kD + (size_t)m0 * kD;
  const bf16_t* Wr = WoB + (size_t)(wave * 96) * (2 * kD);

  v8f acc[6] = {};
  v16bf a_cur = load_a_bf(avB, kD);
  v16bf b_cur[6];
#pragma unroll
  for (int t = 0; t < 6; ++t)
    b_cur[t] = load_b_bf(Wr + (size_t)(t * 16) * (2 * kD), 2 * kD);

  for (int k = 0; k < 2 * kD; k += 32) {
    const bool more = (k + 32) < 2 * kD;
    v16bf a_nxt = a_cur;
    v16bf b_nxt[6];
    if (more) {
      const int kn = k + 32;
      const bf16_t* src = (kn < kD) ? (avB + kn) : (alB + (kn - kD));
      a_nxt = load_a_bf(src, kD);
#pragma unroll
      for (int t = 0; t < 6; ++t)
        b_nxt[t] = load_b_bf(Wr + (size_t)(t * 16) * (2 * kD) + kn, 2 * kD);
    }
#pragma unroll
    for (int t = 0; t < 6; ++t) acc[t] = wmma_bf16(a_cur, b_cur[t], acc[t]);
    a_cur = a_nxt;
    if (more) {
#pragma unroll
      for (int t = 0; t < 6; ++t) b_cur[t] = b_nxt[t];
    }
  }

  const int nl = lane & 15, rbase = (lane >> 4) * 8;
  float s1[8] = {}, s2[8] = {};
#pragma unroll
  for (int t = 0; t < 6; ++t) {
    const int n = wave * 96 + t * 16 + nl;
    const float bb = bo[n];
#pragma unroll
    for (int r = 0; r < 8; ++r) {
      float h = acc[t][r] + bb;
      acc[t][r] = h;
      s1[r] += h;
      s2[r] += h * h;
    }
  }
#pragma unroll
  for (int r = 0; r < 8; ++r)
    for (int s = 1; s < 16; s <<= 1) {
      s1[r] += __shfl_xor(s1[r], s, 32);
      s2[r] += __shfl_xor(s2[r], s, 32);
    }

  __shared__ float p1[8][16];
  __shared__ float p2[8][16];
  if (lane == 0)  for (int r = 0; r < 8; ++r) { p1[wave][r]     = s1[r]; p2[wave][r]     = s2[r]; }
  if (lane == 16) for (int r = 0; r < 8; ++r) { p1[wave][8 + r] = s1[r]; p2[wave][8 + r] = s2[r]; }
  __syncthreads();

  float mu[8], rs[8];
#pragma unroll
  for (int r = 0; r < 8; ++r) {
    float a = 0.f, q = 0.f;
    for (int w = 0; w < 8; ++w) { a += p1[w][rbase + r]; q += p2[w][rbase + r]; }
    const float m = a * (1.0f / kD);
    const float v = q * (1.0f / kD) - m * m;
    mu[r] = m;
    rs[r] = rsqrtf(v + kLnEps);
  }

  float* ob = out + ((size_t)b * kS + m0) * kD;
#pragma unroll
  for (int t = 0; t < 6; ++t) {
    const int n = wave * 96 + t * 16 + nl;
    const float g = gamma[n], be = beta[n];
#pragma unroll
    for (int r = 0; r < 8; ++r) {
      float y = (acc[t][r] - mu[r]) * rs[r] * g + be;
      ob[(size_t)(rbase + r) * kD + n] = fmaxf(y, 0.0f);
    }
  }
}

// ---- launch --------------------------------------------------------------

extern "C" void kernel_launch(void* const* d_in, const int* in_sizes, int n_in,
                              void* d_out, int out_size, void* d_ws, size_t ws_size,
                              hipStream_t stream) {
  (void)in_sizes; (void)n_in; (void)out_size; (void)ws_size;
  const float* vis   = (const float*)d_in[0];
  const float* lang  = (const float*)d_in[1];
  const float* Wv    = (const float*)d_in[2];
  const float* bv    = (const float*)d_in[3];
  const float* Wl    = (const float*)d_in[4];
  const float* bl    = (const float*)d_in[5];
  const float* Wo    = (const float*)d_in[6];
  const float* bo    = (const float*)d_in[7];
  const float* gamma = (const float*)d_in[8];
  const float* beta  = (const float*)d_in[9];
  float* out = (float*)d_out;

  bf16_t* ws = (bf16_t*)d_ws;
  size_t off = 0;
  bf16_t* WvB  = ws + off; off += (size_t)kD * kD;
  bf16_t* WlB  = ws + off; off += (size_t)kD * kD;
  bf16_t* WoBf = ws + off; off += (size_t)kD * 2 * kD;
  bf16_t* visB = ws + off; off += (size_t)kB * kS * kD;
  bf16_t* lanB = ws + off; off += (size_t)kB * kS * kD;
  bf16_t* vp   = ws + off; off += (size_t)kB * kS * kD;
  bf16_t* lp   = ws + off; off += (size_t)kB * kS * kD;
  bf16_t* av   = ws + off; off += (size_t)kB * kS * kD;
  bf16_t* al   = ws + off; off += (size_t)kB * kS * kD;
  bf16_t* attn = ws + off; off += (size_t)kB * kS * kS;

  {
    int n = kD * kD;
    k_f32_to_bf16<<<(n + 255) / 256, 256, 0, stream>>>(Wv, WvB, n);
    k_f32_to_bf16<<<(n + 255) / 256, 256, 0, stream>>>(Wl, WlB, n);
    int n2 = kD * 2 * kD;
    k_f32_to_bf16<<<(n2 + 255) / 256, 256, 0, stream>>>(Wo, WoBf, n2);
    int n3 = kB * kS * kD;
    k_f32_to_bf16<<<(n3 + 255) / 256, 256, 0, stream>>>(vis, visB, n3);
    k_f32_to_bf16<<<(n3 + 255) / 256, 256, 0, stream>>>(lang, lanB, n3);
  }
  k_proj<<<dim3(384, kB, 2), 128, 0, stream>>>(visB, lanB, WvB, WlB, bv, bl, vp, lp);
  k_sim_softmax<<<dim3(kS / 16, kB), 256, 0, stream>>>(vp, lp, attn);
  k_aligned_vision<<<dim3(32 * 12, kB), 128, 0, stream>>>(attn, vp, av);
  k_aligned_language<<<dim3(32 * 12, kB), 128, 0, stream>>>(attn, lp, al);
  k_head_ln_relu<<<dim3(kS / 16, kB), 256, 0, stream>>>(av, al, WoBf, bo, gamma, beta, out);
}
